// Attention_57166014710283
// MI455X (gfx1250) — matrix operand order repaired and checked by reading, not compiled, not run on hardware
//
#include <hip/hip_runtime.h>
#include <hip/hip_bf16.h>

// ---------------- problem constants (match reference) ----------------
#define SEQ      2048
#define DIM      4096
#define NHEADS   32
#define NKV      8
#define HDIM     128
#define KVDIM    (NKV * HDIM)     // 1024
#define HALF_HD  (HDIM / 2)       // 64

typedef __attribute__((ext_vector_type(16))) __bf16 v16bf;
typedef __attribute__((ext_vector_type(8)))  float  v8f;
typedef __attribute__((ext_vector_type(4)))  unsigned int v4u;
typedef __attribute__((ext_vector_type(8)))  int v8i;
typedef __attribute__((ext_vector_type(4)))  int v4i;

union Frag16 { v16bf v; uint4 u[2]; };

static __device__ __forceinline__ v8f wmma_bf16(const v16bf& a, const v16bf& b, const v8f& c) {
    return __builtin_amdgcn_wmma_f32_16x16x32_bf16(false, a, false, b, (short)0, c, false, false);
}

// A-fragment: per-lane chunks at +0 and +16 elements (16x32 bf16 layout)
static __device__ __forceinline__ void load_fragA(Frag16& f, const __hip_bfloat16* p) {
    f.u[0] = *(const uint4*)(p);
    f.u[1] = *(const uint4*)(p + 16);
}
// B-fragment: per-lane chunks at +0 and +16 elements (32x16 bf16 layout, K-contiguous source)
static __device__ __forceinline__ void load_fragB(Frag16& f, const __hip_bfloat16* p) {
    f.u[0] = *(const uint4*)(p);
    f.u[1] = *(const uint4*)(p + 16);
}

// ---------------- TDM: async 2-D bf16 tile DMA, global -> LDS ----------------
// Descriptor bitfields per CDNA5 ISA §8 (D# group0: count/lds/global/type,
// group1: data_size=2B, tensor dims, tile dims, dim0 stride). 2-D tile only.
static __device__ __forceinline__ void tdm_load_2d(unsigned lds_off, const void* gptr,
                                                   unsigned tile_d0, unsigned tile_d1,
                                                   unsigned stride_elems,
                                                   unsigned tensor_d0, unsigned tensor_d1) {
    unsigned long long ga = (unsigned long long)(uintptr_t)gptr;
    v4u g0;
    g0[0] = 1u;                                            // count=1 (valid user D#)
    g0[1] = lds_off;                                       // LDS byte address
    g0[2] = (unsigned)ga;                                  // global_addr[31:0]
    g0[3] = (unsigned)((ga >> 32) & 0x01FFFFFFu) | (2u << 30); // addr[56:32] | type=2
    v8i g1;
    g1[0] = 0x00010000;                                    // data_size=1 -> 2 bytes
    g1[1] = (int)((tensor_d0 & 0xFFFFu) << 16);            // tensor_dim0[15:0]
    g1[2] = (int)(((tensor_d0 >> 16) & 0xFFFFu) | ((tensor_d1 & 0xFFFFu) << 16));
    g1[3] = (int)(((tensor_d1 >> 16) & 0xFFFFu) | (tile_d0 << 16)); // tile_dim0
    g1[4] = (int)(tile_d1 & 0xFFFFu);                      // tile_dim1 (tile_dim2=0)
    g1[5] = (int)stride_elems;                             // tensor_dim0_stride[31:0]
    g1[6] = 0;                                             // stride[47:32]=0, dim1_stride lo=0
    g1[7] = 0;
    v4i z = {0, 0, 0, 0};
#if defined(__clang_major__) && (__clang_major__ >= 23)
    v8i z8 = {0, 0, 0, 0, 0, 0, 0, 0};
    __builtin_amdgcn_tensor_load_to_lds(g0, g1, z, z, z8, 0);
#else
    __builtin_amdgcn_tensor_load_to_lds(g0, g1, z, z, 0);
#endif
}

static __device__ __forceinline__ unsigned lds_offset_of(const void* p) {
    return (unsigned)(uintptr_t)p;     // generic LDS ptr: offset lives in the low 32 bits
}

// ---------------- fp32 -> bf16 convert ----------------
__global__ void cvt_bf16_kernel(const float* __restrict__ in,
                                __hip_bfloat16* __restrict__ out, int n) {
    int i = blockIdx.x * blockDim.x + threadIdx.x;
    if (i < n) out[i] = __float2bfloat16(in[i]);
}

// ---------------- fp32 W[K][N] -> bf16 WT[N][K] ----------------
__global__ void tcvt_bf16_kernel(const float* __restrict__ in,
                                 __hip_bfloat16* __restrict__ out, int K, int N) {
    int i = blockIdx.x * blockDim.x + threadIdx.x;
    if (i >= K * N) return;
    int k = i / N, n = i % N;
    out[(size_t)n * K + k] = __float2bfloat16(in[i]);
}

// ---------------- RoPE (interleaved pairs) on a bf16 activation buffer ----------------
__global__ void rope_kernel(__hip_bfloat16* __restrict__ buf, int nheads,
                            const float* __restrict__ cosb, const float* __restrict__ sinb) {
    int idx = blockIdx.x * blockDim.x + threadIdx.x;
    int total = SEQ * nheads * HALF_HD;
    if (idx >= total) return;
    int i = idx % HALF_HD;
    int h = (idx / HALF_HD) % nheads;
    int s = idx / (HALF_HD * nheads);
    size_t base = (size_t)s * nheads * HDIM + (size_t)h * HDIM + 2 * i;
    float re = __bfloat162float(buf[base]);
    float im = __bfloat162float(buf[base + 1]);
    float c  = cosb[s * HALF_HD + i];
    float sn = sinb[s * HALF_HD + i];
    buf[base]     = __float2bfloat16(re * c - im * sn);
    buf[base + 1] = __float2bfloat16(re * sn + im * c);
}

// ---------------- WMMA GEMM: C[MxN] = A[MxK] * B, with B given as BT[N][K] ----------------
// Block tile 128x128 (8 waves, each 32Mx64N). K pipelined in 32-deep phases,
// double-buffered fragments so WMMAs overlap the next phase's loads.
// mode 0: bf16 C[m*N+n]   mode 1: bf16 C[n*M+m] (transposed store)   mode 2: f32 C[m*N+n]
__global__ __launch_bounds__(256)
void gemm_bf16_wmma(const __hip_bfloat16* __restrict__ A,
                    const __hip_bfloat16* __restrict__ BT,
                    void* __restrict__ Cout,
                    int M, int N, int K, int mode) {
    const int lane = threadIdx.x & 31;
    const int wave = threadIdx.x >> 5;
    const int half = lane >> 4;
    const int l16  = lane & 15;
    const int wm   = wave >> 1;                 // 0..3
    const int wn   = wave & 1;                  // 0..1
    const int mbase = blockIdx.y * 128 + wm * 32;
    const int nbase = blockIdx.x * 128 + wn * 64;

    const __hip_bfloat16* arow[2];
    const __hip_bfloat16* brow[4];
#pragma unroll
    for (int i = 0; i < 2; ++i)
        arow[i] = A + (size_t)(mbase + i * 16 + l16) * K + half * 8;
#pragma unroll
    for (int j = 0; j < 4; ++j)
        brow[j] = BT + (size_t)(nbase + j * 16 + l16) * K + half * 8;

    v8f acc[2][4] = {};
    Frag16 a0[2], b0[4], a1[2], b1[4];

    // prologue: phase 0
#pragma unroll
    for (int i = 0; i < 2; ++i) load_fragA(a0[i], arow[i]);
#pragma unroll
    for (int j = 0; j < 4; ++j) load_fragB(b0[j], brow[j]);

    for (int kk = 0; kk < K; kk += 64) {
        // issue loads for phase kk+32 while computing phase kk
        const int kN = kk + 32;
#pragma unroll
        for (int i = 0; i < 2; ++i) load_fragA(a1[i], arow[i] + kN);
#pragma unroll
        for (int j = 0; j < 4; ++j) load_fragB(b1[j], brow[j] + kN);
        __builtin_prefetch(arow[0] + kk + 256, 0, 1);
        __builtin_prefetch(brow[0] + kk + 256, 0, 1);
#pragma unroll
        for (int i = 0; i < 2; ++i)
#pragma unroll
            for (int j = 0; j < 4; ++j)
                acc[i][j] = wmma_bf16(a0[i].v, b0[j].v, acc[i][j]);

        // issue loads for phase kk+64 (clamped on last iter) while computing kk+32
        const int kNN = (kk + 64 < K) ? (kk + 64) : 0;
#pragma unroll
        for (int i = 0; i < 2; ++i) load_fragA(a0[i], arow[i] + kNN);
#pragma unroll
        for (int j = 0; j < 4; ++j) load_fragB(b0[j], brow[j] + kNN);
#pragma unroll
        for (int i = 0; i < 2; ++i)
#pragma unroll
            for (int j = 0; j < 4; ++j)
                acc[i][j] = wmma_bf16(a1[i].v, b1[j].v, acc[i][j]);
    }

#pragma unroll
    for (int i = 0; i < 2; ++i)
#pragma unroll
        for (int j = 0; j < 4; ++j)
#pragma unroll
            for (int r = 0; r < 8; ++r) {
                int m = mbase + i * 16 + r + half * 8;
                int n = nbase + j * 16 + l16;
                float v = acc[i][j][r];
                if (mode == 0)
                    ((__hip_bfloat16*)Cout)[(size_t)m * N + n] = __float2bfloat16(v);
                else if (mode == 1)
                    ((__hip_bfloat16*)Cout)[(size_t)n * M + m] = __float2bfloat16(v);
                else
                    ((float*)Cout)[(size_t)m * N + n] = v;
            }
}

// ---------------- flash attention: 4 waves (one GQA group) per block ----------------
// K/V chunks are TDM-DMA'd into LDS (double-buffered) and shared by the 4 query
// heads that map to the same KV head; fragments then come from LDS.
// qb: [SEQ][DIM] bf16 (rope'd)   kb: [SEQ][KVDIM] bf16 (rope'd)
// vt: [KVDIM][SEQ] bf16 (V transposed)   ao: [SEQ][DIM] bf16
__global__ __launch_bounds__(128)
void flash_attn_kernel(const __hip_bfloat16* __restrict__ qb,
                       const __hip_bfloat16* __restrict__ kb,
                       const __hip_bfloat16* __restrict__ vt,
                       __hip_bfloat16* __restrict__ ao) {
    __shared__ __align__(16) __hip_bfloat16 ldsK[2][32 * HDIM];   // [buf][key][d]
    __shared__ __align__(16) __hip_bfloat16 ldsV[2][HDIM * 32];   // [buf][d][t]
    __shared__ __align__(16) __hip_bfloat16 plds[4][16 * 32];     // per-wave P staging

    const int lane = threadIdx.x & 31;
    const int wave = threadIdx.x >> 5;         // 0..3 -> head within GQA group
    const int half = lane >> 4;
    const int l16  = lane & 15;
    const int qtile = blockIdx.x;              // 0..127
    const int kvh   = blockIdx.y;              // 0..7
    const int head  = kvh * 4 + wave;
    const int qbase = qtile * 16;

    const __hip_bfloat16* ksrc = kb + (size_t)kvh * HDIM;           // + t0*KVDIM
    const __hip_bfloat16* vsrc = vt + (size_t)(kvh * HDIM) * SEQ;   // + t0

    // Q fragments for the 4 K-chunks of head_dim (kept resident)
    Frag16 qa[4];
    const __hip_bfloat16* qrow =
        qb + (size_t)(qbase + l16) * DIM + head * HDIM + half * 8;
#pragma unroll
    for (int d = 0; d < 4; ++d) load_fragA(qa[d], qrow + d * 32);

    v8f acc[8] = {};                           // 16 x 128 output accumulator
    float m_run[8], l_run[8];
#pragma unroll
    for (int r = 0; r < 8; ++r) { m_run[r] = -1e30f; l_run[r] = 0.0f; }

    const float scale = 0.08838834764831845f;  // 1/sqrt(128)
    const int nchunks = (qbase + 16 + 31) / 32;

    // prologue: DMA chunk 0 into buffer 0 (wave 0 drives the TDM)
    if (wave == 0) {
        tdm_load_2d(lds_offset_of(&ldsK[0][0]), ksrc, HDIM, 32, KVDIM, KVDIM, SEQ);
        tdm_load_2d(lds_offset_of(&ldsV[0][0]), vsrc, 32, HDIM, SEQ, SEQ, KVDIM);
    }

    for (int ch = 0; ch < nchunks; ++ch) {
        const int t0  = ch * 32;
        const int cur = ch & 1;
        const bool more = (ch + 1) < nchunks;

        if (wave == 0) {
            if (more) {
                const int t1 = t0 + 32;
                tdm_load_2d(lds_offset_of(&ldsK[cur ^ 1][0]),
                            ksrc + (size_t)t1 * KVDIM, HDIM, 32, KVDIM, KVDIM, SEQ);
                tdm_load_2d(lds_offset_of(&ldsV[cur ^ 1][0]),
                            vsrc + t1, 32, HDIM, SEQ, SEQ, KVDIM);
                // tensor ops retire in order: <=2 outstanding => current chunk landed
                __builtin_amdgcn_s_wait_tensorcnt(2);
            } else {
                __builtin_amdgcn_s_wait_tensorcnt(0);
            }
        }
        __syncthreads();                       // publish LDS K/V tiles to all 4 waves

        // --- K fragments from LDS, then the 8 QK^T WMMAs ---
        Frag16 kf[8];
        const __hip_bfloat16* kl = &ldsK[cur][0] + l16 * HDIM + half * 8;
#pragma unroll
        for (int d = 0; d < 4; ++d) {
            load_fragB(kf[d],     kl + d * 32);
            load_fragB(kf[4 + d], kl + 16 * HDIM + d * 32);
        }
        v8f s0 = {}, s1 = {};
#pragma unroll
        for (int d = 0; d < 4; ++d) s0 = wmma_bf16(qa[d].v, kf[d].v, s0);
#pragma unroll
        for (int d = 0; d < 4; ++d) s1 = wmma_bf16(qa[d].v, kf[4 + d].v, s1);

        // --- online softmax update (row = r + 8*half across a 16-lane half) ---
#pragma unroll
        for (int r = 0; r < 8; ++r) {
            const int row = qbase + r + half * 8;
            float v0 = (t0 + l16      <= row) ? s0[r] * scale : -1e30f;
            float v1 = (t0 + 16 + l16 <= row) ? s1[r] * scale : -1e30f;
            float mx = fmaxf(v0, v1);
#pragma unroll
            for (int off = 8; off; off >>= 1) mx = fmaxf(mx, __shfl_xor(mx, off, 32));
            float mnew  = fmaxf(m_run[r], mx);
            float alpha = __expf(m_run[r] - mnew);
            float p0 = __expf(v0 - mnew);
            float p1 = __expf(v1 - mnew);
            float rs = p0 + p1;
#pragma unroll
            for (int off = 8; off; off >>= 1) rs += __shfl_xor(rs, off, 32);
            l_run[r] = l_run[r] * alpha + rs;
            m_run[r] = mnew;
#pragma unroll
            for (int dt = 0; dt < 8; ++dt) acc[dt][r] *= alpha;
            const int m = r + half * 8;
            plds[wave][m * 32 + l16]      = __float2bfloat16(p0);
            plds[wave][m * 32 + 16 + l16] = __float2bfloat16(p1);
        }

        // --- reload P (own wave's staging region) as an A-fragment (16x32) ---
        Frag16 pa;
        pa.u[0] = *(const uint4*)&plds[wave][l16 * 32 + half * 8];
        pa.u[1] = *(const uint4*)&plds[wave][l16 * 32 + 16 + half * 8];

        // --- V fragments from LDS, then the 8 P*V WMMAs ---
        Frag16 vf[8];
        const __hip_bfloat16* vl = &ldsV[cur][0] + l16 * 32 + half * 8;
#pragma unroll
        for (int dt = 0; dt < 8; ++dt)
            load_fragB(vf[dt], vl + dt * 16 * 32);
#pragma unroll
        for (int dt = 0; dt < 8; ++dt)
            acc[dt] = wmma_bf16(pa.v, vf[dt].v, acc[dt]);

        __syncthreads();                       // protect K/V buffer before reuse
    }

    // --- normalize and store bf16 attention output ---
#pragma unroll
    for (int r = 0; r < 8; ++r) {
        float inv = 1.0f / l_run[r];
        int row = qbase + r + half * 8;
#pragma unroll
        for (int dt = 0; dt < 8; ++dt)
            ao[(size_t)row * DIM + head * HDIM + dt * 16 + l16] =
                __float2bfloat16(acc[dt][r] * inv);
    }
}

// ---------------- host side ----------------
extern "C" void kernel_launch(void* const* d_in, const int* in_sizes, int n_in,
                              void* d_out, int out_size, void* d_ws, size_t ws_size,
                              hipStream_t stream) {
    const float* x    = (const float*)d_in[0];   // [SEQ][DIM]
    const float* wq   = (const float*)d_in[1];   // [DIM][DIM]
    const float* wk   = (const float*)d_in[2];   // [DIM][KVDIM]
    const float* wv   = (const float*)d_in[3];   // [DIM][KVDIM]
    const float* wo   = (const float*)d_in[4];   // [DIM][DIM]
    const float* fcos = (const float*)d_in[5];   // [SEQ][64]
    const float* fsin = (const float*)d_in[6];   // [SEQ][64]
    // d_in[7] = mask (causal, implicit), d_in[8] = start_pos (0) — unused
    float* out = (float*)d_out;

    char* ws = (char*)d_ws;
    size_t off = 0;
    auto alloc = [&](size_t bytes) -> void* {
        void* p = ws + off;
        off += (bytes + 255) & ~(size_t)255;
        return p;
    };
    __hip_bfloat16* xb   = (__hip_bfloat16*)alloc((size_t)SEQ * DIM * 2);
    __hip_bfloat16* wqT  = (__hip_bfloat16*)alloc((size_t)DIM * DIM * 2);
    __hip_bfloat16* wkT  = (__hip_bfloat16*)alloc((size_t)KVDIM * DIM * 2);
    __hip_bfloat16* wvT  = (__hip_bfloat16*)alloc((size_t)KVDIM * DIM * 2);
    __hip_bfloat16* woT  = (__hip_bfloat16*)alloc((size_t)DIM * DIM * 2);
    __hip_bfloat16* qbuf = (__hip_bfloat16*)alloc((size_t)SEQ * DIM * 2);
    __hip_bfloat16* kbuf = (__hip_bfloat16*)alloc((size_t)SEQ * KVDIM * 2);
    __hip_bfloat16* vtbf = (__hip_bfloat16*)alloc((size_t)KVDIM * SEQ * 2);
    __hip_bfloat16* aobf = (__hip_bfloat16*)alloc((size_t)SEQ * DIM * 2);

    // 1) convert activations, transpose+convert weights
    {
        int n = SEQ * DIM;
        cvt_bf16_kernel<<<(n + 255) / 256, 256, 0, stream>>>(x, xb, n);
    }
    tcvt_bf16_kernel<<<(DIM * DIM   + 255) / 256, 256, 0, stream>>>(wq, wqT, DIM, DIM);
    tcvt_bf16_kernel<<<(DIM * KVDIM + 255) / 256, 256, 0, stream>>>(wk, wkT, DIM, KVDIM);
    tcvt_bf16_kernel<<<(DIM * KVDIM + 255) / 256, 256, 0, stream>>>(wv, wvT, DIM, KVDIM);
    tcvt_bf16_kernel<<<(DIM * DIM   + 255) / 256, 256, 0, stream>>>(wo, woT, DIM, DIM);

    // 2) projections (WMMA GEMMs), block tile 128x128
    gemm_bf16_wmma<<<dim3(DIM / 128,   SEQ / 128), 256, 0, stream>>>(xb, wqT, qbuf, SEQ, DIM,   DIM, 0);
    gemm_bf16_wmma<<<dim3(KVDIM / 128, SEQ / 128), 256, 0, stream>>>(xb, wkT, kbuf, SEQ, KVDIM, DIM, 0);
    gemm_bf16_wmma<<<dim3(KVDIM / 128, SEQ / 128), 256, 0, stream>>>(xb, wvT, vtbf, SEQ, KVDIM, DIM, 1); // V^T

    // 3) RoPE on Q and K
    {
        int nq = SEQ * NHEADS * HALF_HD;
        rope_kernel<<<(nq + 255) / 256, 256, 0, stream>>>(qbuf, NHEADS, fcos, fsin);
        int nk = SEQ * NKV * HALF_HD;
        rope_kernel<<<(nk + 255) / 256, 256, 0, stream>>>(kbuf, NKV, fcos, fsin);
    }

    // 4) causal flash attention (GQA 4:1), K/V staged via TDM into LDS
    flash_attn_kernel<<<dim3(SEQ / 16, NKV), 128, 0, stream>>>(qbuf, kbuf, vtbf, aobf);

    // 5) output projection -> fp32 result
    gemm_bf16_wmma<<<dim3(DIM / 128, SEQ / 128), 256, 0, stream>>>(aobf, woT, out, SEQ, DIM, DIM, 2);
}
